// MHA_19516331393724
// MI455X (gfx1250) — matrix operand order repaired and checked by reading, not compiled, not run on hardware
//
#include <hip/hip_runtime.h>

// ---------------------------------------------------------------------------
// Types
// ---------------------------------------------------------------------------
typedef __attribute__((ext_vector_type(4)))  unsigned short us4;
typedef __attribute__((ext_vector_type(8)))  __bf16         v8bf;
typedef __attribute__((ext_vector_type(16))) __bf16         v16bf;
typedef __attribute__((ext_vector_type(8)))  float          v8f;
typedef __attribute__((ext_vector_type(4)))  unsigned int   u32x4;
typedef __attribute__((ext_vector_type(8)))  int            i32x8;
typedef __attribute__((ext_vector_type(4)))  int            i32x4;

#define DMODEL 1024
#define NHEADS 16
#define DK     64
#define BATCH  4
#define SEQ    2048

// fp32 -> bf16 round-to-nearest-even
__device__ __forceinline__ unsigned short f2bf_u(float f) {
    union { float f; unsigned u; } c; c.f = f;
    unsigned u = c.u + 0x7FFFu + ((c.u >> 16) & 1u);
    return (unsigned short)(u >> 16);
}

__device__ __forceinline__ v16bf cat8(v8bf lo, v8bf hi) {
    return __builtin_shufflevector(lo, hi, 0,1,2,3,4,5,6,7,8,9,10,11,12,13,14,15);
}

// A-matrix 16x32 bf16 fragment, wave32 layout (ISA 7.12.2):
//   lane<16 : row = lane,    halves = K[0..7],  K[16..23]
//   lane>=16: row = lane-16, halves = K[8..15], K[24..31]
__device__ __forceinline__ v16bf load_afrag(const unsigned short* p) {
    v8bf lo = *reinterpret_cast<const v8bf*>(p);
    v8bf hi = *reinterpret_cast<const v8bf*>(p + 16);
    return cat8(lo, hi);
}

// B-matrix 32x16 bf16 fragment:
//   lane<16 : col = lane,    halves = K[0..7],  K[16..23]
//   lane>=16: col = lane-16, halves = K[8..15], K[24..31]
__device__ __forceinline__ v16bf load_bfrag(const unsigned short* p) {
    v8bf lo = *reinterpret_cast<const v8bf*>(p);
    v8bf hi = *reinterpret_cast<const v8bf*>(p + 16);
    return cat8(lo, hi);
}

__device__ __forceinline__ v8f wmma_bf16(v16bf a, v16bf b, v8f c) {
    return __builtin_amdgcn_wmma_f32_16x16x32_bf16(false, a, false, b,
                                                   (short)0, c, false, false);
}

// Async global -> LDS copy, 16B per lane (GLOBAL_LOAD_ASYNC_TO_LDS_B128).
__device__ __forceinline__ void async_copy_b128(unsigned ldsOff, const void* gsrc) {
    asm volatile("global_load_async_to_lds_b128 %0, %1, off"
                 :: "v"(ldsOff), "v"((unsigned long long)(size_t)gsrc)
                 : "memory");
}
__device__ __forceinline__ void wait_asynccnt0() {
    asm volatile("s_wait_asynccnt 0x0" ::: "memory");
}
__device__ __forceinline__ unsigned lds_off(const void* p) {
    return (unsigned)(size_t)p;   // low 32 bits of generic shared addr = LDS offset
}

// DPP row_ror reduction step (stays within a 16-lane DPP row; wave32 has two).
// ctrl: 0x120 + n  (row_ror:n)
template<int CTRL>
__device__ __forceinline__ float dpp_ror_f(float x) {
    int xi = __builtin_bit_cast(int, x);
    int r  = __builtin_amdgcn_update_dpp(xi, xi, CTRL, 0xf, 0xf, true);
    return __builtin_bit_cast(float, r);
}
// max across the 16 lanes of this lane's DPP row (pure VALU, no LDS round-trip)
__device__ __forceinline__ float rowmax16(float x) {
    x = fmaxf(x, dpp_ror_f<0x121>(x));   // ror 1
    x = fmaxf(x, dpp_ror_f<0x122>(x));   // ror 2
    x = fmaxf(x, dpp_ror_f<0x124>(x));   // ror 4
    x = fmaxf(x, dpp_ror_f<0x128>(x));   // ror 8
    return x;
}

// ---------------------------------------------------------------------------
// GEMM: Out[M,N] = A[M,K] @ W[N,K]^T + bias[N]
//   A_BF16  : A is bf16 (staged by TDM tensor_load_to_lds with LDS padding)
//             else fp32 (converted to bf16 while staging through VGPRs)
//   OUT_F32 : Out is fp32 else bf16
// Block: 256 threads (8 waves), tile 128x128, BK = 32.
// ---------------------------------------------------------------------------
template<bool A_BF16, bool OUT_F32>
__global__ __launch_bounds__(256, 2)
void gemm_bias_kernel(const void* __restrict__ Ain,
                      const float* __restrict__ W,
                      const float* __restrict__ bias,
                      void* __restrict__ Out,
                      int M, int N, int K)
{
    __shared__ unsigned short ldsA[128 * 40];   // 128 rows x 32 (+8 pad) bf16
    __shared__ unsigned short ldsW[128 * 40];

    const int tid   = threadIdx.x;
    const int lane  = tid & 31;
    const int wave  = tid >> 5;
    const int half  = (lane >> 4) & 1;
    const int l15   = lane & 15;
    const int waveM = wave & 3;
    const int waveN = wave >> 2;
    const int mBase = blockIdx.y * 128;
    const int nBase = blockIdx.x * 128;

    v8f acc[2][4] = {};

    const int kTiles = K >> 5;
    for (int kb = 0; kb < kTiles; ++kb) {
        __syncthreads();   // previous tile consumed by all waves

        // ---- stage A tile (128x32) into LDS as bf16 ----
        if constexpr (A_BF16) {
            // Tensor Data Mover: one wave issues a 2D tiled DMA with LDS
            // padding producing the 40-half padded rows directly.
            if (wave == 0) {
                const unsigned short* A = (const unsigned short*)Ain;
                unsigned long long ga =
                    (unsigned long long)(size_t)(A + (size_t)mBase * K + kb * 32);
                u32x4 g0;
                g0[0] = 1u;                                  // count = 1
                g0[1] = lds_off(&ldsA[0]);                   // lds_addr
                g0[2] = (unsigned)ga;                        // global_addr lo
                g0[3] = (unsigned)(ga >> 32) | (2u << 30);   // addr hi | type=2
                i32x8 g1;
                g1[0] = (1 << 16)      // data_size = 2 bytes
                      | (1 << 20)      // pad_enable
                      | (3 << 22)      // pad_interval: every 16 DWORDs (64B row)
                      | (3 << 25);     // pad_amount: 4 DWORDs (8 halves)
                g1[1] = (int)((unsigned)(K & 0xFFFF) << 16);           // dim0 lo16
                g1[2] = (int)(((unsigned)K >> 16) |
                              ((unsigned)(M & 0xFFFF) << 16));         // dim0 hi, dim1 lo
                g1[3] = (int)((((unsigned)M >> 16) & 0xFFFFu) |
                              (32u << 16));                            // dim1 hi, tile_dim0=32
                g1[4] = 128;                                           // tile_dim1=128
                g1[5] = K;                                             // dim0_stride lo32
                g1[6] = 0;
                g1[7] = 0;
                i32x4 g2 = {0, 0, 0, 0};
                i32x4 g3 = {0, 0, 0, 0};
                i32x8 g4 = {0, 0, 0, 0, 0, 0, 0, 0};
                __builtin_amdgcn_tensor_load_to_lds(g0, g1, g2, g3, g4, 0);
                __builtin_amdgcn_s_wait_tensorcnt(0);
            }
        } else {
            const float* A = (const float*)Ain;
            #pragma unroll
            for (int i = 0; i < 4; ++i) {
                int id  = tid + i * 256;
                int row = id >> 3, c4 = id & 7;
                const float* src = A + (size_t)(mBase + row) * K + kb * 32 + c4 * 4;
                if (i == 0 && kb + 1 < kTiles) __builtin_prefetch(src + 32, 0, 0);
                float4 v = *reinterpret_cast<const float4*>(src);
                us4 o;
                o[0] = f2bf_u(v.x); o[1] = f2bf_u(v.y);
                o[2] = f2bf_u(v.z); o[3] = f2bf_u(v.w);
                *reinterpret_cast<us4*>(&ldsA[row * 40 + c4 * 4]) = o;
            }
        }
        // ---- stage W tile (128 rows of N, 32 of K), fp32 -> bf16 ----
        #pragma unroll
        for (int i = 0; i < 4; ++i) {
            int id  = tid + i * 256;
            int row = id >> 3, c4 = id & 7;
            const float* src = W + (size_t)(nBase + row) * K + kb * 32 + c4 * 4;
            if (i == 0 && kb + 1 < kTiles) __builtin_prefetch(src + 32, 0, 0);
            float4 v = *reinterpret_cast<const float4*>(src);
            us4 o;
            o[0] = f2bf_u(v.x); o[1] = f2bf_u(v.y);
            o[2] = f2bf_u(v.z); o[3] = f2bf_u(v.w);
            *reinterpret_cast<us4*>(&ldsW[row * 40 + c4 * 4]) = o;
        }
        __syncthreads();

        // ---- fragments + WMMA ----
        v16bf afrag[2];
        #pragma unroll
        for (int mt = 0; mt < 2; ++mt) {
            int row = waveM * 32 + mt * 16 + l15;
            afrag[mt] = load_afrag(&ldsA[row * 40 + half * 8]);
        }
        #pragma unroll
        for (int nt = 0; nt < 4; ++nt) {
            int col = waveN * 64 + nt * 16 + l15;
            v16bf bfrag = load_bfrag(&ldsW[col * 40 + half * 8]);
            #pragma unroll
            for (int mt = 0; mt < 2; ++mt)
                acc[mt][nt] = wmma_bf16(afrag[mt], bfrag, acc[mt][nt]);
        }
    }

    // ---- epilogue: bias + store (C layout: lane half -> M+8, elem r -> M+r)
    #pragma unroll
    for (int nt = 0; nt < 4; ++nt) {
        int n = nBase + waveN * 64 + nt * 16 + l15;
        float bv = bias[n];
        #pragma unroll
        for (int mt = 0; mt < 2; ++mt) {
            #pragma unroll
            for (int r = 0; r < 8; ++r) {
                int m = mBase + waveM * 32 + mt * 16 + r + half * 8;
                float val = acc[mt][nt][r] + bv;
                if (OUT_F32)
                    ((float*)Out)[(size_t)m * N + n] = val;
                else
                    ((unsigned short*)Out)[(size_t)m * N + n] = f2bf_u(val);
            }
        }
    }
}

// ---------------------------------------------------------------------------
// Flash attention over bf16 Q/K/V in (B,S,D) layout; head h = cols [h*64,h*64+64)
// Block: 128 threads (4 waves), each wave owns 16 query rows; 64-key chunks.
// V chunks double-buffered via async global->LDS copies.  Softmax row-sums are
// produced by the WMMA pipe itself (P @ ones); row-max uses DPP row_ror.
// ---------------------------------------------------------------------------
__global__ __launch_bounds__(128, 2)
void attn_kernel(const unsigned short* __restrict__ Q,
                 const unsigned short* __restrict__ Kb,
                 const unsigned short* __restrict__ V,
                 const int* __restrict__ mask,
                 unsigned short* __restrict__ Oa)
{
    __shared__ unsigned short Plds[4][16][72];   // per-wave P tile (16 q x 64 keys)
    __shared__ unsigned short Vt[64][72];        // V^T chunk: [dk][key]
    __shared__ unsigned short Vrow[2][64][64];   // async landing buffers [key][dk]

    const int tid  = threadIdx.x;
    const int lane = tid & 31;
    const int wave = tid >> 5;
    const int half = (lane >> 4) & 1;
    const int l15  = lane & 15;
    const int rOff = half * 8;
    const int bh   = blockIdx.y;
    const int b    = bh >> 4;
    const int h    = bh & 15;
    const int qBase = blockIdx.x * 64;
    const int NC   = SEQ / 64;

    // all-ones 32x16 B fragment (bf16 1.0 = 0x3F80) for WMMA row-sum
    __bf16 one;
    { unsigned short hbits = 0x3F80; __builtin_memcpy(&one, &hbits, 2); }
    v16bf onesB;
    #pragma unroll
    for (int i = 0; i < 16; ++i) onesB[i] = one;

    // Q fragments (A layout): contiguous per-lane chunks of this lane's Q row
    v16bf qf[2];
    {
        const size_t qrow = (size_t)(b * SEQ + qBase + wave * 16 + l15) * DMODEL + h * DK;
        #pragma unroll
        for (int kt = 0; kt < 2; ++kt)
            qf[kt] = load_afrag(Q + qrow + kt * 32 + half * 8);
    }

    // issue async copy of V chunk 0 (4 x 16B per thread)
    {
        #pragma unroll
        for (int i = 0; i < 4; ++i) {
            int id  = tid + i * 128;
            int key = id >> 3, c = id & 7;     // 8 x 16B chunks per 64-half row
            const void* src = V + (size_t)(b * SEQ + key) * DMODEL + h * DK + c * 8;
            async_copy_b128(lds_off(&Vrow[0][key][c * 8]), src);
        }
    }

    float mrow[8];
    #pragma unroll
    for (int r = 0; r < 8; ++r) mrow[r] = -1e30f;
    v8f oacc[4] = {};
    v8f sacc = {};     // running row-sums (every lane holds its rows' sums)

    for (int kc = 0; kc < NC; ++kc) {
        const int keyBase = kc * 64;
        const int buf = kc & 1;

        wait_asynccnt0();   // this chunk's V has landed (issued last iteration)
        __syncthreads();    // everyone's copies visible; Vt free for overwrite

        // ---- issue async copy of NEXT chunk's V into the other buffer ----
        if (kc + 1 < NC) {
            #pragma unroll
            for (int i = 0; i < 4; ++i) {
                int id  = tid + i * 128;
                int key = id >> 3, c = id & 7;
                const void* src =
                    V + (size_t)(b * SEQ + keyBase + 64 + key) * DMODEL + h * DK + c * 8;
                async_copy_b128(lds_off(&Vrow[buf ^ 1][key][c * 8]), src);
            }
        }

        // ---- transpose Vrow[buf] -> Vt (LDS -> LDS) ----
        #pragma unroll
        for (int i = 0; i < 8; ++i) {
            int id  = tid + i * 128;
            int key = id >> 4, c4 = id & 15;
            us4 v = *reinterpret_cast<const us4*>(&Vrow[buf][key][c4 * 4]);
            #pragma unroll
            for (int j = 0; j < 4; ++j) Vt[c4 * 4 + j][key] = v[j];
        }

        // ---- scores: S = Q @ K^T  (B-frag = contiguous slice of a K row) ----
        v8f s[4] = {};
        #pragma unroll
        for (int nt = 0; nt < 4; ++nt) {
            const size_t krow =
                (size_t)(b * SEQ + keyBase + nt * 16 + l15) * DMODEL + h * DK;
            if (kc + 1 < NC)
                __builtin_prefetch(Kb + krow + (size_t)64 * DMODEL, 0, 0);
            #pragma unroll
            for (int kt = 0; kt < 2; ++kt) {
                v16bf bf = load_bfrag(Kb + krow + kt * 32 + half * 8);
                s[nt] = wmma_bf16(qf[kt], bf, s[nt]);
            }
        }

        // ---- scale + mask ----
        #pragma unroll
        for (int nt = 0; nt < 4; ++nt) {
            int mk = mask[b * SEQ + keyBase + nt * 16 + l15];
            float madd = mk ? 0.f : -1e30f;
            #pragma unroll
            for (int r = 0; r < 8; ++r) s[nt][r] = s[nt][r] * 0.125f + madd;
        }

        // ---- online softmax: row-max via DPP (VALU only) ----
        float alpha[8];
        #pragma unroll
        for (int r = 0; r < 8; ++r) {
            float rmax = fmaxf(fmaxf(s[0][r], s[1][r]), fmaxf(s[2][r], s[3][r]));
            rmax = rowmax16(rmax);
            float mnew = fmaxf(mrow[r], rmax);
            alpha[r] = __expf(mrow[r] - mnew);
            mrow[r]  = mnew;
        }
        #pragma unroll
        for (int nt = 0; nt < 4; ++nt)
            #pragma unroll
            for (int r = 0; r < 8; ++r)
                s[nt][r] = __expf(s[nt][r] - mrow[r]);

        // rescale running accumulators (row sums rescale identically)
        #pragma unroll
        for (int r = 0; r < 8; ++r) {
            sacc[r] *= alpha[r];
            #pragma unroll
            for (int nt = 0; nt < 4; ++nt) oacc[nt][r] *= alpha[r];
        }

        // ---- P (C layout) -> LDS -> A layout ----
        #pragma unroll
        for (int nt = 0; nt < 4; ++nt)
            #pragma unroll
            for (int r = 0; r < 8; ++r)
                Plds[wave][r + rOff][nt * 16 + l15] = f2bf_u(s[nt][r]);
        __syncthreads();   // Vt + P visible to all fragment readers

        // ---- O += P @ V ; row-sums += P @ ones (matrix pipe, no shuffles) ----
        v16bf pf[2];
        #pragma unroll
        for (int kt = 0; kt < 2; ++kt)
            pf[kt] = load_afrag(&Plds[wave][l15][kt * 32 + half * 8]);
        #pragma unroll
        for (int kt = 0; kt < 2; ++kt)
            sacc = wmma_bf16(pf[kt], onesB, sacc);
        #pragma unroll
        for (int nt = 0; nt < 4; ++nt) {
            #pragma unroll
            for (int kt = 0; kt < 2; ++kt) {
                v16bf bf = load_bfrag(&Vt[nt * 16 + l15][kt * 32 + half * 8]);
                oacc[nt] = wmma_bf16(pf[kt], bf, oacc[nt]);
            }
        }
    }

    // ---- normalize & store (bf16, (B,S,D) layout with head offset) ----
    #pragma unroll
    for (int r = 0; r < 8; ++r) {
        float inv = 1.0f / sacc[r];
        size_t m = (size_t)(b * SEQ + qBase + wave * 16 + r + rOff);
        #pragma unroll
        for (int nt = 0; nt < 4; ++nt)
            Oa[m * DMODEL + h * DK + nt * 16 + l15] = f2bf_u(oacc[nt][r] * inv);
    }
}

// ---------------------------------------------------------------------------
// Launch.  d_in: q,k,v,mask,w_q,b_q,w_k,b_k,w_v,b_v,w_o,b_o
// ---------------------------------------------------------------------------
extern "C" void kernel_launch(void* const* d_in, const int* in_sizes, int n_in,
                              void* d_out, int out_size, void* d_ws, size_t ws_size,
                              hipStream_t stream)
{
    (void)in_sizes; (void)n_in; (void)out_size; (void)ws_size;

    const float* q    = (const float*)d_in[0];
    const float* k    = (const float*)d_in[1];
    const float* v    = (const float*)d_in[2];
    const int*   mask = (const int*)  d_in[3];
    const float* w_q  = (const float*)d_in[4];
    const float* b_q  = (const float*)d_in[5];
    const float* w_k  = (const float*)d_in[6];
    const float* b_k  = (const float*)d_in[7];
    const float* w_v  = (const float*)d_in[8];
    const float* b_v  = (const float*)d_in[9];
    const float* w_o  = (const float*)d_in[10];
    const float* b_o  = (const float*)d_in[11];
    float* out = (float*)d_out;

    const size_t nHalf = (size_t)BATCH * SEQ * DMODEL; // 8,388,608 elems
    unsigned short* Qbf = (unsigned short*)d_ws;
    unsigned short* Kbf = Qbf + nHalf;
    unsigned short* Vbf = Kbf + nHalf;
    unsigned short* Abf = Vbf + nHalf;     // 64 MB of workspace total

    const int M = BATCH * SEQ;             // 8192
    const int N = DMODEL, Kd = DMODEL;     // 1024
    dim3 gGrid(N / 128, M / 128);          // (8, 64)

    gemm_bias_kernel<false, false><<<gGrid, 256, 0, stream>>>(q, w_q, b_q, Qbf, M, N, Kd);
    gemm_bias_kernel<false, false><<<gGrid, 256, 0, stream>>>(k, w_k, b_k, Kbf, M, N, Kd);
    gemm_bias_kernel<false, false><<<gGrid, 256, 0, stream>>>(v, w_v, b_v, Vbf, M, N, Kd);

    dim3 aGrid(SEQ / 64, BATCH * NHEADS); // (32, 64)
    attn_kernel<<<aGrid, 128, 0, stream>>>(Qbf, Kbf, Vbf, mask, Abf);

    gemm_bias_kernel<true, true><<<gGrid, 256, 0, stream>>>(Abf, w_o, b_o, out, M, N, Kd);
}
